// MultiQueryAttention_67430986547257
// MI455X (gfx1250) — matrix operand order repaired and checked by reading, not compiled, not run on hardware
//
#include <hip/hip_runtime.h>

typedef __attribute__((ext_vector_type(16))) __bf16 v16bf;
typedef __attribute__((ext_vector_type(8)))  __bf16 v8bf;
typedef __attribute__((ext_vector_type(8)))  float  v8f;
typedef __attribute__((ext_vector_type(4)))  int    v4i;

#define DMODEL 1024
#define SEQ    2048
#define NBATCH 2
#define NHEAD  32
#define DHEAD  32
#define MTOT   (NBATCH * SEQ)   // 4096

// ---------------------------------------------------------------------------
// CDNA5 async global->LDS copy (ASYNCcnt) with compile-safe fallback.
// Builtin signature (from hipcc diagnostics): (v4i AS1*, v4i AS3*, Imm, Imm)
// ---------------------------------------------------------------------------
#if defined(__gfx1250__) &&                                                    \
    __has_builtin(__builtin_amdgcn_global_load_async_to_lds_b128) &&           \
    __has_builtin(__builtin_amdgcn_s_wait_asynccnt)
#define HAVE_ASYNC_LDS 1
#else
#define HAVE_ASYNC_LDS 0
#endif

__device__ __forceinline__ void async_copy16(const __bf16* g, __bf16* l) {
#if HAVE_ASYNC_LDS
  __builtin_amdgcn_global_load_async_to_lds_b128(
      (__attribute__((address_space(1))) v4i*)g,
      (__attribute__((address_space(3))) v4i*)l, 0, 0);
#else
  *(v8bf*)l = *(const v8bf*)g;   // synchronous bounce through registers
#endif
}

template <int N>
__device__ __forceinline__ void wait_async() {
#if HAVE_ASYNC_LDS
  __builtin_amdgcn_s_wait_asynccnt(N);   // N is a constant-expression here
#endif
}

// ---------------------------------------------------------------------------
// Fragment helpers (wave32); A/B use this code's contiguous k order, not the ISA table order.
// A (16x32 bf16, MxK): lane l -> row m = l&15, half = l>>4.
//   element i holds K = half*16 + i
// B (32x16 bf16, KxN): lane l -> col n = l&15; element j holds K = half*16 + j
// C/D (16x16 f32): lane l -> col n = l&15; acc[r] is row m = r + (l>>4)*8.
// ---------------------------------------------------------------------------
__device__ __forceinline__ v16bf load_afrag(const __bf16* rowp, int half) {
  union { v16bf v; v8bf h[2]; } u;
  u.h[0] = *(const v8bf*)(rowp + half * 16);
  u.h[1] = *(const v8bf*)(rowp + half * 16 + 8);
  return u.v;
}
__device__ __forceinline__ v16bf load_bfrag(const __bf16* p) {
  return *(const v16bf*)p;   // 16 contiguous bf16, 32B aligned
}
__device__ __forceinline__ v16bf load_bfrag16(const __bf16* p) {
  // same as load_bfrag but only 16B alignment guaranteed (LDS padded rows)
  union { v16bf v; v8bf h[2]; } u;
  u.h[0] = *(const v8bf*)(p);
  u.h[1] = *(const v8bf*)(p + 8);
  return u.v;
}
__device__ __forceinline__ v8f wmma_bf16(v16bf a, v16bf b, v8f c) {
  return __builtin_amdgcn_wmma_f32_16x16x32_bf16(false, a, false, b, (short)0, c,
                                                 false, false);
}

// ---------------------------------------------------------------------------
// fp32 -> bf16 elementwise convert
// ---------------------------------------------------------------------------
__global__ void cvt_f32_bf16(const float* __restrict__ src,
                             __bf16* __restrict__ dst, int n) {
  int i = blockIdx.x * blockDim.x + threadIdx.x;
  int stride = gridDim.x * blockDim.x;
  for (; i < n; i += stride) dst[i] = (__bf16)src[i];
}

// ---------------------------------------------------------------------------
// LDS-tiled transpose + convert: src fp32 [K x N] -> dst bf16 [N x K]
// blockDim = (32, 8)
// ---------------------------------------------------------------------------
__global__ void transpose_cvt(const float* __restrict__ src,
                              __bf16* __restrict__ dst, int K, int N) {
  __shared__ float tile[32][33];
  int n = blockIdx.x * 32 + threadIdx.x;
#pragma unroll
  for (int j = 0; j < 4; ++j) {
    int k = blockIdx.y * 32 + threadIdx.y + j * 8;
    if (k < K && n < N) tile[threadIdx.y + j * 8][threadIdx.x] = src[(size_t)k * N + n];
  }
  __syncthreads();
  int k2 = blockIdx.y * 32 + threadIdx.x;
#pragma unroll
  for (int j = 0; j < 4; ++j) {
    int n2 = blockIdx.x * 32 + threadIdx.y + j * 8;
    if (k2 < K && n2 < N)
      dst[(size_t)n2 * K + k2] = (__bf16)tile[threadIdx.x][threadIdx.y + j * 8];
  }
}

// ---------------------------------------------------------------------------
// Small-N GEMM (direct from global): out[M x N] = A[M x K] * Bt[N x K]^T + bias
// Wave computes a 16x32 tile. Used for K/V projections (N = 32).
// mode 1: bf16 row-major out; mode 2: bf16 "V-transposed" out (N==32)
// ---------------------------------------------------------------------------
__global__ void gemm_bf16(const __bf16* __restrict__ A, const __bf16* __restrict__ Bt,
                          const float* __restrict__ bias, void* __restrict__ out,
                          int M, int N, int K, int mode) {
  const int wave = threadIdx.x >> 5;
  const int lane = threadIdx.x & 31;
  const int lr = lane & 15;
  const int half = lane >> 4;
  const int m0 = (blockIdx.y * 8 + wave) * 16;
  const int n0 = blockIdx.x * 32;
  if (m0 >= M || n0 >= N) return;  // wave-uniform

  const __bf16* arow  = A  + (size_t)(m0 + lr) * K;
  const __bf16* b0row = Bt + (size_t)(n0 + lr) * K;
  const __bf16* b1row = Bt + (size_t)(n0 + 16 + lr) * K;

  v8f acc0 = {}, acc1 = {};
  for (int k0 = 0; k0 < K; k0 += 32) {
    __builtin_prefetch(arow + k0 + 256, 0, 1);   // global_prefetch_b8
    v16bf a  = load_afrag(arow + k0, half);
    v16bf b0 = load_bfrag(b0row + k0 + half * 16);
    v16bf b1 = load_bfrag(b1row + k0 + half * 16);
    acc0 = wmma_bf16(a, b0, acc0);
    acc1 = wmma_bf16(a, b1, acc1);
  }

  const float bia0 = bias[n0 + lr];
  const float bia1 = bias[n0 + 16 + lr];
#pragma unroll
  for (int r = 0; r < 8; ++r) {
    int m = m0 + r + half * 8;
    float v0 = acc0[r] + bia0;
    float v1 = acc1[r] + bia1;
    if (mode == 1) {
      __bf16* o = (__bf16*)out;
      o[(size_t)m * N + n0 + lr]      = (__bf16)v0;
      o[(size_t)m * N + n0 + 16 + lr] = (__bf16)v1;
    } else {
      __bf16* o = (__bf16*)out;   // N == 32 here: out[(b*32+n)*SEQ + s]
      int bb = m >> 11, s = m & (SEQ - 1);
      o[((size_t)bb * DHEAD + n0 + lr) * SEQ + s]      = (__bf16)v0;
      o[((size_t)bb * DHEAD + n0 + 16 + lr) * SEQ + s] = (__bf16)v1;
    }
  }
}

// ---------------------------------------------------------------------------
// Big GEMM with async-LDS weight staging.
// out[M x N] = A[M x K] * Bt[N x K]^T + bias
// block = 256 thr (8 waves). Block tile: M=128, N=64. K chunk = 64, double
// buffered in LDS via GLOBAL_LOAD_ASYNC_TO_LDS_B128 (ASYNCcnt). Each wave
// computes 16x64 (4 accumulators) -> 4 WMMAs per A-fragment.
// mode 0: fp32 out (+bias); mode 1: bf16 out (+bias).
// ---------------------------------------------------------------------------
#define BSTRIDE 72   // 64 + 8 pad elems -> 144B row stride, spreads LDS banks
__global__ void gemm_bf16_lds(const __bf16* __restrict__ A,
                              const __bf16* __restrict__ Bt,
                              const float* __restrict__ bias,
                              void* __restrict__ out,
                              int M, int N, int K, int mode) {
  __shared__ __bf16 bufB[2][64][BSTRIDE];   // 2 x 64 rows x 64 elems (+pad)
  const int tid  = threadIdx.x;
  const int wave = tid >> 5;
  const int lane = tid & 31;
  const int lr   = lane & 15;
  const int half = lane >> 4;
  const int m0 = blockIdx.y * 128 + wave * 16;
  const int bn = blockIdx.x * 64;

  const __bf16* arow = A + (size_t)(m0 + lr) * K;

  // B-tile copy: 64 rows x 128B = 8 KB = 512 x 16B chunks, 2 per thread
  const int c0 = tid * 2;
  const int r0 = c0 >> 3;              // row in 64-row tile
  const int o0 = (c0 & 7) * 8;         // element offset in row
  const int r1 = (c0 + 1) >> 3;
  const int o1 = ((c0 + 1) & 7) * 8;
  const __bf16* bsrc0 = Bt + (size_t)(bn + r0) * K + o0;
  const __bf16* bsrc1 = Bt + (size_t)(bn + r1) * K + o1;

  const int nk = K >> 6;
  // prologue: fill stage 0
  async_copy16(bsrc0, &bufB[0][r0][o0]);
  async_copy16(bsrc1, &bufB[0][r1][o1]);

  v8f acc[4] = {};
  for (int kc = 0; kc < nk; ++kc) {
    const int st = kc & 1;
    if (kc + 1 < nk) {                  // kick off next stage, then wait for
      const int k1 = (kc + 1) << 6;     // everything older than it (in-order)
      async_copy16(bsrc0 + k1, &bufB[st ^ 1][r0][o0]);
      async_copy16(bsrc1 + k1, &bufB[st ^ 1][r1][o1]);
      wait_async<2>();
    } else {
      wait_async<0>();
    }
    __syncthreads();                    // stage `st` visible to all waves

    __builtin_prefetch(arow + (kc << 6) + 512, 0, 1);
#pragma unroll
    for (int sub = 0; sub < 2; ++sub) {
      const int k0 = (kc << 6) + sub * 32;
      v16bf a = load_afrag(arow + k0, half);
#pragma unroll
      for (int j = 0; j < 4; ++j) {
        v16bf b = load_bfrag16(&bufB[st][j * 16 + lr][sub * 32 + half * 16]);
        acc[j] = wmma_bf16(a, b, acc[j]);
      }
    }
    __syncthreads();                    // reads done before buffer is refilled
  }

#pragma unroll
  for (int j = 0; j < 4; ++j) {
    const int col = bn + j * 16 + lr;
    const float bia = bias[col];
#pragma unroll
    for (int r = 0; r < 8; ++r) {
      int m = m0 + r + half * 8;
      float v = acc[j][r] + bia;
      if (mode == 0) ((float*)out)[(size_t)m * N + col] = v;
      else           ((__bf16*)out)[(size_t)m * N + col] = (__bf16)v;
    }
  }
}

// ---------------------------------------------------------------------------
// Flash MQA: one wave per (batch, head a, 16-query block).
// q  : bf16 [MTOT x DMODEL]  (head a uses cols a*32 .. a*32+31)
// k  : bf16 [MTOT x DHEAD]   row-major (shared head)
// vT : bf16 [NBATCH x DHEAD x SEQ] (V transposed -> contiguous B-frags)
// attn: bf16 [MTOT x DMODEL], head a writes cols a*32 .. a*32+31
// ---------------------------------------------------------------------------
#define PSTRIDE 40   // 32 + 8 pad elems -> 80B row stride for bank spread
__global__ void mqa_flash(const __bf16* __restrict__ q, const __bf16* __restrict__ k,
                          const __bf16* __restrict__ vT, __bf16* __restrict__ attn) {
  __shared__ __bf16 Pbuf[8][16 * PSTRIDE];   // per-wave P tile
  const int wave = threadIdx.x >> 5;
  const int lane = threadIdx.x & 31;
  const int lr = lane & 15;
  const int half = lane >> 4;

  int gid = blockIdx.x * 8 + wave;       // 0 .. 8191
  int b   = gid >> 12;                   // 4096 row-blocks per batch
  int rem = gid & 4095;
  int a   = rem >> 7;                    // head 0..31
  int q0  = (rem & 127) << 4;            // query row block start

  const float scale = 0.17677669529663687f;  // 1/sqrt(32)

  const __bf16* qrow = q + ((size_t)(b * SEQ + q0 + lr) * DMODEL + a * DHEAD);
  v16bf qf = load_afrag(qrow, half);

  float mi[8], li[8];
  v8f o0 = {}, o1 = {};
#pragma unroll
  for (int r = 0; r < 8; ++r) { mi[r] = -__builtin_inff(); li[r] = 0.f; }

  __bf16* P = Pbuf[wave];
  const __bf16* kbase  = k  + (size_t)b * SEQ * DHEAD;
  const __bf16* v0base = vT + (size_t)(b * DHEAD + lr) * SEQ;
  const __bf16* v1base = vT + (size_t)(b * DHEAD + 16 + lr) * SEQ;

  for (int t = 0; t < SEQ; t += 32) {
    // scores: Q(16x32) @ K^T(32 x 32keys) -> two 16x16 f32 tiles
    v16bf kf0 = load_bfrag(kbase + (size_t)(t + lr) * DHEAD + half * 16);
    v16bf kf1 = load_bfrag(kbase + (size_t)(t + 16 + lr) * DHEAD + half * 16);
    v8f s0 = {}, s1 = {};
    s0 = wmma_bf16(qf, kf0, s0);
    s1 = wmma_bf16(qf, kf1, s1);

    float mv[8], rs[8], alpha[8];
#pragma unroll
    for (int r = 0; r < 8; ++r) {
      float x0 = s0[r] * scale, x1 = s1[r] * scale;
      s0[r] = x0; s1[r] = x1;
      mv[r] = fmaxf(x0, x1);
    }
#pragma unroll
    for (int off = 1; off < 16; off <<= 1)
#pragma unroll
      for (int r = 0; r < 8; ++r)
        mv[r] = fmaxf(mv[r], __shfl_xor(mv[r], off, 32));

#pragma unroll
    for (int r = 0; r < 8; ++r) {
      float mn = fmaxf(mi[r], mv[r]);
      alpha[r] = __expf(mi[r] - mn);
      mi[r] = mn;
      float p0 = __expf(s0[r] - mn);
      float p1 = __expf(s1[r] - mn);
      rs[r] = p0 + p1;
      int row = r + half * 8;
      P[row * PSTRIDE + lr]      = (__bf16)p0;   // C-layout -> LDS
      P[row * PSTRIDE + 16 + lr] = (__bf16)p1;
    }
#pragma unroll
    for (int off = 1; off < 16; off <<= 1)
#pragma unroll
      for (int r = 0; r < 8; ++r)
        rs[r] += __shfl_xor(rs[r], off, 32);

    // same-wave LDS is in order; make the cross-lane hand-off explicit
#if __has_builtin(__builtin_amdgcn_s_wait_dscnt)
    __builtin_amdgcn_s_wait_dscnt(0);
#else
    __syncthreads();
#endif

    // re-read P in A-fragment layout
    union { v16bf v; v8bf h[2]; } pu;
    pu.h[0] = *(const v8bf*)(P + lr * PSTRIDE + half * 16);
    pu.h[1] = *(const v8bf*)(P + lr * PSTRIDE + half * 16 + 8);

    v16bf vf0 = load_bfrag(v0base + t + half * 16);
    v16bf vf1 = load_bfrag(v1base + t + half * 16);

#pragma unroll
    for (int r = 0; r < 8; ++r) {
      li[r] = li[r] * alpha[r] + rs[r];
      o0[r] *= alpha[r];
      o1[r] *= alpha[r];
    }
    o0 = wmma_bf16(pu.v, vf0, o0);
    o1 = wmma_bf16(pu.v, vf1, o1);
  }

#pragma unroll
  for (int r = 0; r < 8; ++r) {
    float inv = 1.0f / li[r];
    int row = b * SEQ + q0 + r + half * 8;
    size_t base = (size_t)row * DMODEL + a * DHEAD;
    attn[base + lr]      = (__bf16)(o0[r] * inv);
    attn[base + 16 + lr] = (__bf16)(o1[r] * inv);
  }
}

// ---------------------------------------------------------------------------
extern "C" void kernel_launch(void* const* d_in, const int* in_sizes, int n_in,
                              void* d_out, int out_size, void* d_ws, size_t ws_size,
                              hipStream_t stream) {
  (void)in_sizes; (void)n_in; (void)out_size; (void)ws_size;
  const float* x  = (const float*)d_in[0];
  const float* Wq = (const float*)d_in[1];
  const float* bq = (const float*)d_in[2];
  const float* Wk = (const float*)d_in[3];
  const float* bk = (const float*)d_in[4];
  const float* Wv = (const float*)d_in[5];
  const float* bv = (const float*)d_in[6];
  const float* Wo = (const float*)d_in[7];
  const float* bo = (const float*)d_in[8];
  float* out = (float*)d_out;

  // workspace carve-up (~29 MB)
  char* ws = (char*)d_ws;
  size_t o = 0;
  __bf16* x_bf    = (__bf16*)(ws + o); o += (size_t)MTOT * DMODEL * 2;
  __bf16* WqT     = (__bf16*)(ws + o); o += (size_t)DMODEL * DMODEL * 2;
  __bf16* WkT     = (__bf16*)(ws + o); o += (size_t)DHEAD * DMODEL * 2;
  __bf16* WvT     = (__bf16*)(ws + o); o += (size_t)DHEAD * DMODEL * 2;
  __bf16* WoT     = (__bf16*)(ws + o); o += (size_t)DMODEL * DMODEL * 2;
  __bf16* q_bf    = (__bf16*)(ws + o); o += (size_t)MTOT * DMODEL * 2;
  __bf16* k_bf    = (__bf16*)(ws + o); o += (size_t)MTOT * DHEAD * 2;
  __bf16* vT_bf   = (__bf16*)(ws + o); o += (size_t)MTOT * DHEAD * 2;
  __bf16* attn_bf = (__bf16*)(ws + o); o += (size_t)MTOT * DMODEL * 2;

  dim3 tb(32, 8);
  cvt_f32_bf16<<<1024, 256, 0, stream>>>(x, x_bf, MTOT * DMODEL);
  transpose_cvt<<<dim3(32, 32), tb, 0, stream>>>(Wq, WqT, DMODEL, DMODEL);
  transpose_cvt<<<dim3(1, 32),  tb, 0, stream>>>(Wk, WkT, DMODEL, DHEAD);
  transpose_cvt<<<dim3(1, 32),  tb, 0, stream>>>(Wv, WvT, DMODEL, DHEAD);
  transpose_cvt<<<dim3(32, 32), tb, 0, stream>>>(Wo, WoT, DMODEL, DMODEL);

  // Q projection: async-LDS staged GEMM (bf16 out)
  gemm_bf16_lds<<<dim3(16, 32), 256, 0, stream>>>(x_bf, WqT, bq, q_bf,
                                                  MTOT, DMODEL, DMODEL, 1);
  // K/V projections (small N, direct); V stored transposed per batch
  gemm_bf16<<<dim3(1, 32), 256, 0, stream>>>(x_bf, WkT, bk, k_bf,
                                             MTOT, DHEAD, DMODEL, 1);
  gemm_bf16<<<dim3(1, 32), 256, 0, stream>>>(x_bf, WvT, bv, vT_bf,
                                             MTOT, DHEAD, DMODEL, 2);

  // flash attention: 2 batches * 32 heads * 128 query blocks = 8192 waves
  mqa_flash<<<1024, 256, 0, stream>>>(q_bf, k_bf, vT_bf, attn_bf);

  // output projection: async-LDS staged GEMM (fp32 out + bias)
  gemm_bf16_lds<<<dim3(16, 32), 256, 0, stream>>>(attn_bf, WoT, bo, out,
                                                  MTOT, DMODEL, DMODEL, 0);
}
